// CrossCoder_74534862455449
// MI455X (gfx1250) — matrix operand order repaired and checked by reading, not compiled, not run on hardware
//
#include <hip/hip_runtime.h>
#include <hip/hip_bf16.h>

// ---------------------------------------------------------------------------
// CrossCoder forward on gfx1250 (MI455X), bf16 WMMA with f32 accumulation.
//   Encoder: f = relu(x[B,L*D] @ W_enc[L*D,F] + b_enc)      -> bf16 in d_ws
//   Decoder: x_hat[:,l,:] = f @ W_dec[l][F,D] + b_dec[l]    -> f32 in d_out
// Encoder block tile 128x128 (8 waves, 2x4 WMMA tiles each).
// Decoder block tile  64x128 (8 waves, 2x2 WMMA tiles each) for 2x occupancy
// on the K=32768 loop.  f32 operands are converted to bf16 with vector
// fptrunc (-> v_cvt_pk_bf16_f32) while staging into padded double-buffered LDS.
// ---------------------------------------------------------------------------

typedef __attribute__((ext_vector_type(16))) __bf16 v16bf;
typedef __attribute__((ext_vector_type(8)))  __bf16 v8bf;
typedef __attribute__((ext_vector_type(4)))  __bf16 v4bf;
typedef __attribute__((ext_vector_type(8)))  float  v8f;
typedef __attribute__((ext_vector_type(4)))  float  f32x4;

#define BK 32
#define LDT 40                         // padded leading dim (bf16 elems): 80 B = 20 dwords
#define ROW_BYTES (LDT * 2)            // 80
#define B_TILE_BYTES (128 * ROW_BYTES) // B operand tile (col-major, 128 cols)

// 4 f32 -> 4 packed bf16 in one vector fptrunc: backend emits two
// two-operand v_cvt_pk_bf16_f32, result already packed (no mov/shift/or).
static __device__ __forceinline__ uint2 cvt4_bf16(f32x4 v) {
  return __builtin_bit_cast(uint2, __builtin_convertvector(v, v4bf));
}

static __device__ __forceinline__ v16bf mk_frag(v8bf lo, v8bf hi) {
  return __builtin_shufflevector(lo, hi, 0,1,2,3,4,5,6,7,8,9,10,11,12,13,14,15);
}

// One K-step of WMMAs for this wave. LDS layouts (bf16, stride LDT):
//  As: row-major   As[row][k] -> A frag lane: row = off+ln; VGPR0-3 = K[lh*8..+7],
//                                VGPR4-7 = K[lh*8+16..+23]  (ISA 7.12.2, 16-bit A 16x32)
//  Bs: col-major   Bs[col][k] -> B frag lane: col = off+ln; V0-3 = K[lh*8..+7], V4-7 = K[lh*8+16..+23]
template <int MT, int NT>
static __device__ __forceinline__ void wave_mma(const unsigned char* As,
                                                const unsigned char* Bs,
                                                int wmoff, int wnoff, int ln, int lh,
                                                v8f acc[MT][NT]) {
  v16bf a[MT];
#pragma unroll
  for (int mi = 0; mi < MT; ++mi) {
    const unsigned char* pr = As + (unsigned)(wmoff + mi * 16 + ln) * ROW_BYTES;
    v8bf lo = *(const v8bf*)(pr + lh * 16);        // K = lh*8 .. lh*8+7
    v8bf hi = *(const v8bf*)(pr + lh * 16 + 32);   // K = lh*8+16 .. +23
    a[mi] = mk_frag(lo, hi);
  }
#pragma unroll
  for (int ni = 0; ni < NT; ++ni) {
    const unsigned char* pc = Bs + (unsigned)(wnoff + ni * 16 + ln) * ROW_BYTES;
    v8bf lo = *(const v8bf*)(pc + lh * 16);        // K = lh*8 .. lh*8+7
    v8bf hi = *(const v8bf*)(pc + lh * 16 + 32);   // K = lh*8+16 .. +23
    v16bf b = mk_frag(lo, hi);
#pragma unroll
    for (int mi = 0; mi < MT; ++mi)
      acc[mi][ni] = __builtin_amdgcn_wmma_f32_16x16x32_bf16(
          false, a[mi], false, b, (short)0, acc[mi][ni], false, false);
  }
}

// ---------------------------------------------------------------------------
// Encoder: M=1024, K=2048, N=32768.  A=x f32, B=W_enc f32 -> bf16 at staging.
// Block 128x128, waves 4(M) x 2(N), wave tile 32x64.
// ---------------------------------------------------------------------------
__global__ void __launch_bounds__(256)
crosscoder_encode_bf16wmma(const float* __restrict__ x,
                           const float* __restrict__ Wenc,
                           const float* __restrict__ benc,
                           __bf16* __restrict__ f) {
  constexpr int K = 2048, N = 32768, KT = K / BK;
  constexpr int A_TILE = 128 * ROW_BYTES;          // 10240
  constexpr int BUFB   = A_TILE + B_TILE_BYTES;    // 20480
  __shared__ alignas(16) unsigned char lds[2 * BUFB];

  const int tid   = threadIdx.x;
  const int mBase = blockIdx.y * 128;
  const int nBase = blockIdx.x * 128;

  const int ar  = tid >> 1, ac0 = (tid & 1) * 16;        // A: 2 thr/row x 16 f32
  const int bn4 = (tid & 31) * 4, bk0 = (tid >> 5) * 4;  // B: 4 rows x 4 cols

  const float* gA = x + (size_t)(mBase + ar) * K + ac0;
  const float* gB = Wenc + (size_t)bk0 * N + nBase + bn4;

  f32x4 pa[4], pb[4];
  auto ldg = [&](int kt) {
#pragma unroll
    for (int j = 0; j < 4; ++j) pa[j] = *(const f32x4*)(gA + kt * BK + 4 * j);
#pragma unroll
    for (int i = 0; i < 4; ++i)
      pb[i] = *(const f32x4*)(gB + (size_t)kt * BK * N + (size_t)i * N);
  };
  auto sts = [&](unsigned char* base) {
    unsigned char* As = base;
    unsigned char* Bs = base + A_TILE;
#pragma unroll
    for (int j = 0; j < 4; ++j)                    // A row-major
      *(uint2*)(As + (unsigned)ar * ROW_BYTES + (unsigned)(ac0 + 4 * j) * 2) =
          cvt4_bf16(pa[j]);
#pragma unroll
    for (int j = 0; j < 4; ++j) {                  // B transposed -> col-major
      f32x4 c; c[0] = pb[0][j]; c[1] = pb[1][j]; c[2] = pb[2][j]; c[3] = pb[3][j];
      *(uint2*)(Bs + (unsigned)(bn4 + j) * ROW_BYTES + (unsigned)bk0 * 2) =
          cvt4_bf16(c);
    }
  };

  const int lane = tid & 31, wid = tid >> 5;
  const int wmoff = (wid & 3) * 32, wnoff = (wid >> 2) * 64;
  const int ln = lane & 15, lh = lane >> 4;

  v8f acc[2][4] = {};

  ldg(0);
  sts(lds);
  __syncthreads();

  for (int kt = 0; kt < KT; ++kt) {
    if (kt + 1 < KT) ldg(kt + 1);
    unsigned char* cur = lds + (unsigned)(kt & 1) * BUFB;
    wave_mma<2, 4>(cur, cur + A_TILE, wmoff, wnoff, ln, lh, acc);
    if (kt + 1 < KT) sts(lds + (unsigned)((kt + 1) & 1) * BUFB);
    __syncthreads();
  }

  // epilogue: +bias, ReLU, bf16 store. C/D layout: VGPR r -> row r + 8*lh, col ln.
#pragma unroll
  for (int ni = 0; ni < 4; ++ni) {
    const int col = nBase + wnoff + ni * 16 + ln;
    const float bv = benc[col];
#pragma unroll
    for (int mi = 0; mi < 2; ++mi) {
      const int rbase = mBase + wmoff + mi * 16 + lh * 8;
#pragma unroll
      for (int r = 0; r < 8; ++r) {
        float v = acc[mi][ni][r] + bv;
        v = v > 0.0f ? v : 0.0f;
        f[(size_t)(rbase + r) * N + col] = (__bf16)v;
      }
    }
  }
}

// ---------------------------------------------------------------------------
// Decoder: per l: M=1024, K=32768, N=1024.  A=f bf16 (raw copy), B=W_dec f32.
// Block 64x128, waves 2(M) x 4(N), wave tile 32x32 -> 256 workgroups total.
// ---------------------------------------------------------------------------
__global__ void __launch_bounds__(256)
crosscoder_decode_bf16wmma(const __bf16* __restrict__ f,
                           const float* __restrict__ Wdec,
                           const float* __restrict__ bdec,
                           float* __restrict__ out) {
  constexpr int K = 32768, N = 1024, L = 2, KT = K / BK;
  constexpr int A_TILE = 64 * ROW_BYTES;           // 5120
  constexpr int BUFB   = A_TILE + B_TILE_BYTES;    // 15360
  __shared__ alignas(16) unsigned char lds[2 * BUFB];

  const int tid   = threadIdx.x;
  const int l     = blockIdx.z;
  const int mBase = blockIdx.y * 64;
  const int nBase = blockIdx.x * 128;

  const int ar  = tid >> 2, aq = (tid & 3) * 8;    // A: 4 thr/row x 8 bf16 (16 B)
  const int bn4 = (tid & 31) * 4, bk0 = (tid >> 5) * 4;

  const __bf16* gA = f + (size_t)(mBase + ar) * K + aq;
  const float*  gB = Wdec + (size_t)l * K * N + (size_t)bk0 * N + nBase + bn4;

  uint4 pa;
  f32x4 pb[4];
  auto ldg = [&](int kt) {
    pa = *(const uint4*)(gA + kt * BK);
#pragma unroll
    for (int i = 0; i < 4; ++i)
      pb[i] = *(const f32x4*)(gB + (size_t)kt * BK * N + (size_t)i * N);
  };
  auto sts = [&](unsigned char* base) {
    unsigned char* As = base;
    unsigned char* Bs = base + A_TILE;
    *(uint4*)(As + (unsigned)ar * ROW_BYTES + (unsigned)aq * 2) = pa;
#pragma unroll
    for (int j = 0; j < 4; ++j) {
      f32x4 c; c[0] = pb[0][j]; c[1] = pb[1][j]; c[2] = pb[2][j]; c[3] = pb[3][j];
      *(uint2*)(Bs + (unsigned)(bn4 + j) * ROW_BYTES + (unsigned)bk0 * 2) =
          cvt4_bf16(c);
    }
  };

  const int lane = tid & 31, wid = tid >> 5;
  const int wmoff = (wid & 1) * 32, wnoff = (wid >> 1) * 32;
  const int ln = lane & 15, lh = lane >> 4;

  v8f acc[2][2] = {};

  ldg(0);
  sts(lds);
  __syncthreads();

  for (int kt = 0; kt < KT; ++kt) {
    if (kt + 1 < KT) ldg(kt + 1);
    unsigned char* cur = lds + (unsigned)(kt & 1) * BUFB;
    wave_mma<2, 2>(cur, cur + A_TILE, wmoff, wnoff, ln, lh, acc);
    if (kt + 1 < KT) sts(lds + (unsigned)((kt + 1) & 1) * BUFB);
    __syncthreads();
  }

#pragma unroll
  for (int ni = 0; ni < 2; ++ni) {
    const int col = nBase + wnoff + ni * 16 + ln;   // d index
    const float bv = bdec[l * N + col];
#pragma unroll
    for (int mi = 0; mi < 2; ++mi) {
      const int rbase = mBase + wmoff + mi * 16 + lh * 8;
#pragma unroll
      for (int r = 0; r < 8; ++r)
        out[(size_t)(rbase + r) * (L * N) + (size_t)l * N + col] = acc[mi][ni][r] + bv;
    }
  }
}

// ---------------------------------------------------------------------------
extern "C" void kernel_launch(void* const* d_in, const int* in_sizes, int n_in,
                              void* d_out, int out_size, void* d_ws, size_t ws_size,
                              hipStream_t stream) {
  (void)in_sizes; (void)n_in; (void)out_size; (void)ws_size;
  const float* x    = (const float*)d_in[0];   // [1024, 2, 1024]
  const float* Wenc = (const float*)d_in[1];   // [2, 1024, 32768] == [2048, 32768]
  const float* benc = (const float*)d_in[2];   // [32768]
  const float* Wdec = (const float*)d_in[3];   // [2, 32768, 1024]
  const float* bdec = (const float*)d_in[4];   // [2, 1024]
  float* out = (float*)d_out;                  // [1024, 2, 1024]
  __bf16* fbuf = (__bf16*)d_ws;                // f: [1024, 32768] bf16 = 64 MB scratch

  dim3 blk(256, 1, 1);
  dim3 gEnc(32768 / 128, 1024 / 128, 1);       // 256 x 8 = 2048 blocks
  crosscoder_encode_bf16wmma<<<gEnc, blk, 0, stream>>>(x, Wenc, benc, fbuf);
  dim3 gDec(1024 / 128, 1024 / 64, 2);         // 8 x 16 x 2 = 256 blocks
  crosscoder_decode_bf16wmma<<<gDec, blk, 0, stream>>>(fbuf, Wdec, bdec, out);
}
